// GRANMixtureBernoulli_15298673508739
// MI455X (gfx1250) — matrix operand order repaired and checked by reading, not compiled, not run on hardware
//
#include <hip/hip_runtime.h>

#define H 128
#define NNODE 20000
#define M_EDGES 400000
#define EPRED 400000
#define KMIX 20
#define NW 8   // waves (16-row tiles) per block

typedef __bf16 v16bf __attribute__((ext_vector_type(16)));
typedef float  v8f   __attribute__((ext_vector_type(8)));

__device__ __forceinline__ unsigned short f2bf(float f) {
  union { float f; unsigned u; } c; c.f = f;
  unsigned u = c.u;
  return (unsigned short)((u + 0x7FFFu + ((u >> 16) & 1u)) >> 16); // RNE
}

__device__ __forceinline__ v8f wmma_bf16(v16bf a, v16bf b, v8f c) {
  return __builtin_amdgcn_wmma_f32_16x16x32_bf16(false, a, false, b, (short)0, c, false, false);
}

// ---- async copy global(bf16, pre-transposed) -> LDS via CDNA5 async-to-LDS engine
// bytes must be a multiple of 4096; 256 threads x 16B per iteration.
__device__ __forceinline__ void asyncStage16(void* dstLds, const void* src, int bytes, int tid) {
  unsigned l = (unsigned)(size_t)dstLds + (unsigned)(tid * 16);
  unsigned g = (unsigned)(tid * 16);
#pragma unroll
  for (int it = 0; it < 8; ++it) {
    if (it * 4096 < bytes) {
      asm volatile("global_load_async_to_lds_b128 %0, %1, %2"
                   :: "v"(l + (unsigned)(it * 4096)),
                      "v"(g + (unsigned)(it * 4096)),
                      "s"(src)
                   : "memory");
    }
  }
}
#define ASYNC_WAIT() asm volatile("s_wait_asynccnt 0" ::: "memory")

// ---- A fragments for K=128 held in registers, reused across all 8 N-tiles
struct AFrag { union U { v16bf v; uint4 q[2]; } u[4]; };

__device__ __forceinline__ void loadA(AFrag& f, const unsigned short* tile, int lane) {
  const int m = lane & 15, half = lane >> 4;
#pragma unroll
  for (int kk = 0; kk < 4; ++kk) {
    // A frag: e<8 -> K=32kk+8*half+e ; e>=8 -> K=32kk+16+8*half+(e-8)
    f.u[kk].q[0] = *(const uint4*)(tile + m*128 + kk*32 + 8*half);
    f.u[kk].q[1] = *(const uint4*)(tile + m*128 + kk*32 + 16 + 8*half);
  }
}

// D(16x16) = A(16x128) x W(128x16-col-slice n), W transposed [n][k] bf16 in LDS
__device__ __forceinline__ v8f gemmA(const AFrag& f, const unsigned short* wT, int n, int lane) {
  v8f acc = {0.f,0.f,0.f,0.f,0.f,0.f,0.f,0.f};
  const int half = lane >> 4;
#pragma unroll
  for (int kk = 0; kk < 4; ++kk) {
    union { v16bf v; uint4 q[2]; } ub;
    // B frag: e<8 -> K=32kk+8*half+e ; e>=8 -> K=32kk+16+8*half+(e-8) at column n
    ub.q[0] = *(const uint4*)(wT + n*128 + kk*32 + 8*half);
    ub.q[1] = *(const uint4*)(wT + n*128 + kk*32 + 16 + 8*half);
    acc = wmma_bf16(f.u[kk].v, ub.v, acc);
  }
  return acc;
}

// ---------------- weight prep: f32 [k][n] -> bf16 transposed [n][128]
__global__ void gran_prepT(const float* __restrict__ W, unsigned short* __restrict__ out) {
  int i = blockIdx.x*256 + threadIdx.x;   // 16384
  int n = i >> 7, k = i & 127;
  out[n*128 + k] = f2bf(W[k*128 + n]);
}
__global__ void gran_prepT3(const float* __restrict__ W, unsigned short* __restrict__ out) {
  int i = blockIdx.x*256 + threadIdx.x;   // 4096 : 128x20 -> padded [32][128]
  int n = i >> 7, k = i & 127;
  out[n*128 + k] = (n < KMIX) ? f2bf(W[k*KMIX + n]) : (unsigned short)0;
}

// ---------------- input projection: nf[1+row] = A[row]@W_in + b_in; nf[0]=0
__global__ __launch_bounds__(128) void gran_inproj(const float* __restrict__ A,
                                                   const float* __restrict__ Win,
                                                   const float* __restrict__ bin,
                                                   float* __restrict__ nf) {
  int row = blockIdx.x, n = threadIdx.x;
  __shared__ float a[256];
  a[n] = A[(size_t)row*256 + n];
  a[n+128] = A[(size_t)row*256 + 128 + n];
  __syncthreads();
  float acc = bin[n];
  for (int k = 0; k < 256; ++k) acc += a[k] * Win[k*128 + n];
  nf[(size_t)(row+1)*128 + n] = acc;
  if (row == 0) nf[n] = 0.f;
}

__global__ void gran_gather(const float* __restrict__ nf, const int* __restrict__ idx,
                            float* __restrict__ state) {
  int i = blockIdx.x*blockDim.x + threadIdx.x;
  if (i < NNODE*H) state[i] = nf[(size_t)idx[i>>7]*128 + (i & 127)];
}

__global__ void gran_relu(float* __restrict__ p, int n) {
  int i = blockIdx.x*blockDim.x + threadIdx.x;
  if (i < n) p[i] = fmaxf(p[i], 0.f);
}

__global__ void gran_zero(float* __restrict__ p, int n) {
  int i = blockIdx.x*blockDim.x + threadIdx.x;
  if (i < n) p[i] = 0.f;
}

// ---------------- fused edge kernel: msg/att MLPs + sigmoid gate + scatter-add
// LDS: wTA 0, wTB 32768, xrowM 65536, xrowA 67584, biases 69632..71680,
//      diffT 71680, hidT 104448, idx 137216..139264
__global__ __launch_bounds__(256) void gran_edge_kernel(
    const float* __restrict__ state, float* __restrict__ agg,
    const int* __restrict__ edges, const int* __restrict__ att_idx,
    const unsigned short* __restrict__ w1T, const unsigned short* __restrict__ w2T,
    const unsigned short* __restrict__ a1T, const unsigned short* __restrict__ a2T,
    const float* __restrict__ msgW1, const float* __restrict__ msgb1,
    const float* __restrict__ msgb2,
    const float* __restrict__ attW1, const float* __restrict__ attb1,
    const float* __restrict__ attb2) {
  extern __shared__ char smem[];
  unsigned short* wTA   = (unsigned short*)smem;
  unsigned short* wTB   = (unsigned short*)(smem + 32768);
  float* xrowM          = (float*)(smem + 65536);   // W1 rows 128,129,192,193
  float* xrowA          = (float*)(smem + 67584);
  float* b1m            = (float*)(smem + 69632);
  float* b2m            = b1m + 128;
  float* b1a            = b2m + 128;
  float* b2a            = b1a + 128;
  unsigned short* diffT = (unsigned short*)(smem + 71680);
  unsigned short* hidT  = (unsigned short*)(smem + 104448);
  int* srcI = (int*)(smem + 137216);
  int* dstI = srcI + NW*16;
  int* aS   = dstI + NW*16;
  int* aD   = aS   + NW*16;

  const int tid = threadIdx.x, lane = tid & 31, wave = tid >> 5;
  const int half = lane >> 4;
  const int e0 = (blockIdx.x * NW + wave) * 16;
  unsigned short* myD = diffT + wave*2048;
  unsigned short* myH = hidT  + wave*2048;

  // kick off async copy of first weight matrix while staging small tables
  asyncStage16(wTA, w1T, 32768, tid);
  for (int i = tid; i < 512; i += 256) {
    int r = i >> 7, n = i & 127;
    int s = (r < 2) ? (128 + r) : (190 + r);   // rows 128,129,192,193
    xrowM[i] = msgW1[s*128 + n];
    xrowA[i] = attW1[s*128 + n];
  }
  for (int i = tid; i < 128; i += 256) {
    b1m[i] = msgb1[i]; b2m[i] = msgb2[i]; b1a[i] = attb1[i]; b2a[i] = attb2[i];
  }
  if (lane < 16) {
    int s = edges[2*(e0+lane)];
    int d = edges[2*(e0+lane)+1];
    srcI[wave*16+lane] = s; dstI[wave*16+lane] = d;
    aS[wave*16+lane] = att_idx[s] & 1;
    aD[wave*16+lane] = att_idx[d] & 1;
  }
  __syncthreads();
  // diff tile: bf16(state[src]-state[dst])
  for (int q = lane; q < 512; q += 32) {
    int m = q >> 5, c = q & 31;
    float4 a = ((const float4*)(state + (size_t)srcI[wave*16+m]*H))[c];
    float4 b = ((const float4*)(state + (size_t)dstI[wave*16+m]*H))[c];
    ushort4 o;
    o.x = f2bf(a.x-b.x); o.y = f2bf(a.y-b.y); o.z = f2bf(a.z-b.z); o.w = f2bf(a.w-b.w);
    *(ushort4*)(myD + m*128 + 4*c) = o;
  }
  ASYNC_WAIT();
  __syncthreads();          // wTA + diff ready

  AFrag faD; loadA(faD, myD, lane);
  asyncStage16(wTB, w2T, 32768, tid);      // overlap msg-W2 copy with GEMM1
  // ---- msg layer 1: relu(diff@W1 + onehot rows + b1) -> hid
#pragma unroll
  for (int t = 0; t < 8; ++t) {
    int n = t*16 + (lane & 15);
    v8f acc = gemmA(faD, wTA, n, lane);
#pragma unroll
    for (int r = 0; r < 8; ++r) {
      int m = r + 8*half;
      float h = acc[r] + xrowM[aS[wave*16+m]*128+n] + xrowM[(2+aD[wave*16+m])*128+n] + b1m[n];
      myH[m*128+n] = f2bf(fmaxf(h, 0.f));
    }
  }
  ASYNC_WAIT();
  __syncthreads();

  AFrag faH; loadA(faH, myH, lane);
  asyncStage16(wTA, a1T, 32768, tid);      // overlap att-W1 copy with GEMM2
  float msgv[64];
#pragma unroll
  for (int t = 0; t < 8; ++t) {
    int n = t*16 + (lane & 15);
    v8f acc = gemmA(faH, wTB, n, lane);
#pragma unroll
    for (int r = 0; r < 8; ++r) msgv[t*8+r] = acc[r] + b2m[n];
  }
  ASYNC_WAIT();
  __syncthreads();

  asyncStage16(wTB, a2T, 32768, tid);      // overlap att-W2 copy with att GEMM1
  // ---- att layer 1 (reuses faD)
#pragma unroll
  for (int t = 0; t < 8; ++t) {
    int n = t*16 + (lane & 15);
    v8f acc = gemmA(faD, wTA, n, lane);
#pragma unroll
    for (int r = 0; r < 8; ++r) {
      int m = r + 8*half;
      float h = acc[r] + xrowA[aS[wave*16+m]*128+n] + xrowA[(2+aD[wave*16+m])*128+n] + b1a[n];
      myH[m*128+n] = f2bf(fmaxf(h, 0.f));
    }
  }
  ASYNC_WAIT();
  __syncthreads();

  loadA(faH, myH, lane);
  // ---- att layer 2 + sigmoid gate + atomic scatter into agg (L2 f32 atomics)
#pragma unroll
  for (int t = 0; t < 8; ++t) {
    int n = t*16 + (lane & 15);
    v8f acc = gemmA(faH, wTB, n, lane);
#pragma unroll
    for (int r = 0; r < 8; ++r) {
      int m = r + 8*half;
      float a = acc[r] + b2a[n];
      float sg = 1.f / (1.f + __expf(-a));
      atomicAdd(&agg[(size_t)dstI[wave*16+m]*H + n], msgv[t*8+r] * sg);
    }
  }
}

// ---------------- GRU (tiny FLOPs -> plain VALU, coalesced L2 weight reads)
__global__ __launch_bounds__(128) void gran_gru(const float* __restrict__ agg,
                                                float* __restrict__ state,
                                                const float* __restrict__ Wih,
                                                const float* __restrict__ bih,
                                                const float* __restrict__ Whh,
                                                const float* __restrict__ bhh) {
  int node = blockIdx.x, n = threadIdx.x;
  __shared__ float x[128], h[128];
  x[n] = agg[(size_t)node*H + n];
  h[n] = state[(size_t)node*H + n];
  __syncthreads();
  float gi0=bih[n], gi1=bih[n+128], gi2=bih[n+256];
  float gh0=bhh[n], gh1=bhh[n+128], gh2=bhh[n+256];
  for (int k = 0; k < 128; ++k) {
    float xv = x[k], hv = h[k];
    const float* wi = Wih + k*384;
    const float* wh = Whh + k*384;
    gi0 += xv*wi[n]; gi1 += xv*wi[n+128]; gi2 += xv*wi[n+256];
    gh0 += hv*wh[n]; gh1 += hv*wh[n+128]; gh2 += hv*wh[n+256];
  }
  float r  = 1.f/(1.f+__expf(-(gi0+gh0)));
  float z  = 1.f/(1.f+__expf(-(gi1+gh1)));
  float nn = tanhf(gi2 + r*gh2);
  state[(size_t)node*H + n] = (1.f - z)*nn + z*h[n];
}

// ---------------- fused output heads (theta + alpha), WMMA, double-buffered weights
// LDS: wTA 0, wTB 32768, biases 65536..67840, dT 67840, hA 100608, hB 133376,
//      idx 166144..167168
__global__ __launch_bounds__(256) void gran_head_kernel(
    const float* __restrict__ state, const int* __restrict__ gnn,
    const unsigned short* __restrict__ tW1T, const unsigned short* __restrict__ tW2T,
    const unsigned short* __restrict__ tW3T,
    const unsigned short* __restrict__ aW1T, const unsigned short* __restrict__ aW2T,
    const unsigned short* __restrict__ aW3T,
    const float* __restrict__ tb1, const float* __restrict__ tb2, const float* __restrict__ tb3,
    const float* __restrict__ ab1, const float* __restrict__ ab2, const float* __restrict__ ab3,
    float* __restrict__ outT, float* __restrict__ outA) {
  extern __shared__ char smem[];
  unsigned short* wTA = (unsigned short*)smem;
  unsigned short* wTB = (unsigned short*)(smem + 32768);
  float* b1t = (float*)(smem + 65536);
  float* b2t = b1t + 128;
  float* b3t = b2t + 128;       // 32
  float* b1a = b3t + 32;
  float* b2a = b1a + 128;
  float* b3a = b2a + 128;       // 32
  unsigned short* dT = (unsigned short*)(smem + 67840);
  unsigned short* hA = dT + NW*2048;
  unsigned short* hB = hA + NW*2048;
  int* r0I = (int*)(smem + 166144);
  int* r1I = r0I + NW*16;

  const int tid = threadIdx.x, lane = tid & 31, wave = tid >> 5;
  const int half = lane >> 4;
  const int e0 = (blockIdx.x * NW + wave) * 16;
  unsigned short* myD = dT + wave*2048;
  unsigned short* myA = hA + wave*2048;
  unsigned short* myB = hB + wave*2048;

  asyncStage16(wTA, tW1T, 32768, tid);
  for (int i = tid; i < 128; i += 256) {
    b1t[i] = tb1[i]; b2t[i] = tb2[i]; b1a[i] = ab1[i]; b2a[i] = ab2[i];
  }
  for (int i = tid; i < 32; i += 256) {
    b3t[i] = (i < KMIX) ? tb3[i] : 0.f;
    b3a[i] = (i < KMIX) ? ab3[i] : 0.f;
  }
  if (lane < 16) {
    r0I[wave*16+lane] = gnn[2*(e0+lane)];
    r1I[wave*16+lane] = gnn[2*(e0+lane)+1];
  }
  __syncthreads();
  for (int q = lane; q < 512; q += 32) {
    int m = q >> 5, c = q & 31;
    float4 a = ((const float4*)(state + (size_t)r0I[wave*16+m]*H))[c];
    float4 b = ((const float4*)(state + (size_t)r1I[wave*16+m]*H))[c];
    ushort4 o;
    o.x = f2bf(a.x-b.x); o.y = f2bf(a.y-b.y); o.z = f2bf(a.z-b.z); o.w = f2bf(a.w-b.w);
    *(ushort4*)(myD + m*128 + 4*c) = o;
  }
  ASYNC_WAIT();
  __syncthreads();

  AFrag faD; loadA(faD, myD, lane);
  AFrag fh;
  asyncStage16(wTB, tW2T, 32768, tid);
  // theta layer 1
#pragma unroll
  for (int t = 0; t < 8; ++t) {
    int n = t*16 + (lane & 15);
    v8f acc = gemmA(faD, wTA, n, lane);
#pragma unroll
    for (int r = 0; r < 8; ++r) myA[(r+8*half)*128+n] = f2bf(fmaxf(acc[r] + b1t[n], 0.f));
  }
  ASYNC_WAIT();
  __syncthreads();

  loadA(fh, myA, lane);
  asyncStage16(wTA, tW3T, 8192, tid);
  // theta layer 2
#pragma unroll
  for (int t = 0; t < 8; ++t) {
    int n = t*16 + (lane & 15);
    v8f acc = gemmA(fh, wTB, n, lane);
#pragma unroll
    for (int r = 0; r < 8; ++r) myB[(r+8*half)*128+n] = f2bf(fmaxf(acc[r] + b2t[n], 0.f));
  }
  ASYNC_WAIT();
  __syncthreads();

  loadA(fh, myB, lane);
  asyncStage16(wTB, aW1T, 32768, tid);
  // theta layer 3 -> log_theta
#pragma unroll
  for (int t = 0; t < 2; ++t) {
    int n = t*16 + (lane & 15);
    v8f acc = gemmA(fh, wTA, n, lane);
    if (n < KMIX) {
#pragma unroll
      for (int r = 0; r < 8; ++r) outT[(size_t)(e0 + r + 8*half)*KMIX + n] = acc[r] + b3t[n];
    }
  }
  ASYNC_WAIT();
  __syncthreads();

  asyncStage16(wTA, aW2T, 32768, tid);
  // alpha layer 1 (reuses faD)
#pragma unroll
  for (int t = 0; t < 8; ++t) {
    int n = t*16 + (lane & 15);
    v8f acc = gemmA(faD, wTB, n, lane);
#pragma unroll
    for (int r = 0; r < 8; ++r) myA[(r+8*half)*128+n] = f2bf(fmaxf(acc[r] + b1a[n], 0.f));
  }
  ASYNC_WAIT();
  __syncthreads();

  loadA(fh, myA, lane);
  asyncStage16(wTB, aW3T, 8192, tid);
  // alpha layer 2
#pragma unroll
  for (int t = 0; t < 8; ++t) {
    int n = t*16 + (lane & 15);
    v8f acc = gemmA(fh, wTA, n, lane);
#pragma unroll
    for (int r = 0; r < 8; ++r) myB[(r+8*half)*128+n] = f2bf(fmaxf(acc[r] + b2a[n], 0.f));
  }
  ASYNC_WAIT();
  __syncthreads();

  loadA(fh, myB, lane);
  // alpha layer 3 -> raw logits (log_softmax applied by gran_lsm)
#pragma unroll
  for (int t = 0; t < 2; ++t) {
    int n = t*16 + (lane & 15);
    v8f acc = gemmA(fh, wTB, n, lane);
    if (n < KMIX) {
#pragma unroll
      for (int r = 0; r < 8; ++r) outA[(size_t)(e0 + r + 8*half)*KMIX + n] = acc[r] + b3a[n];
    }
  }
}

__global__ void gran_lsm(float* __restrict__ p, int rows) {
  int r = blockIdx.x*blockDim.x + threadIdx.x;
  if (r < rows) {
    float* q = p + (size_t)r*KMIX;
    float mx = q[0];
    for (int i = 1; i < KMIX; ++i) mx = fmaxf(mx, q[i]);
    float s = 0.f;
    for (int i = 0; i < KMIX; ++i) s += __expf(q[i] - mx);
    float lse = mx + __logf(s);
    for (int i = 0; i < KMIX; ++i) q[i] = q[i] - lse;
  }
}

extern "C" void kernel_launch(void* const* d_in, const int* in_sizes, int n_in,
                              void* d_out, int out_size, void* d_ws, size_t ws_size,
                              hipStream_t stream) {
  (void)in_sizes; (void)n_in; (void)out_size; (void)ws_size;
  const float* A_pad  = (const float*)d_in[0];
  const int*   edges  = (const int*)d_in[1];
  const int*   gnn    = (const int*)d_in[2];
  const int*   nif    = (const int*)d_in[3];
  const int*   attIdx = (const int*)d_in[4];
  const float* W_in   = (const float*)d_in[5];
  const float* b_in   = (const float*)d_in[6];
  const float* msgW1  = (const float*)d_in[7];
  const float* msgb1  = (const float*)d_in[8];
  const float* msgW2  = (const float*)d_in[9];
  const float* msgb2  = (const float*)d_in[10];
  const float* attW1  = (const float*)d_in[11];
  const float* attb1  = (const float*)d_in[12];
  const float* attW2  = (const float*)d_in[13];
  const float* attb2  = (const float*)d_in[14];
  const float* gWih   = (const float*)d_in[15];
  const float* gbih   = (const float*)d_in[16];
  const float* gWhh   = (const float*)d_in[17];
  const float* gbhh   = (const float*)d_in[18];
  const float* tW1 = (const float*)d_in[19];
  const float* tb1 = (const float*)d_in[20];
  const float* tW2 = (const float*)d_in[21];
  const float* tb2 = (const float*)d_in[22];
  const float* tW3 = (const float*)d_in[23];
  const float* tb3 = (const float*)d_in[24];
  const float* aW1 = (const float*)d_in[25];
  const float* ab1 = (const float*)d_in[26];
  const float* aW2 = (const float*)d_in[27];
  const float* ab2 = (const float*)d_in[28];
  const float* aW3 = (const float*)d_in[29];
  const float* ab3 = (const float*)d_in[30];

  float* ws    = (float*)d_ws;
  float* nf    = ws;                          // 4097*128 f32
  float* state = nf + 4097*128;               // 20000*128 f32
  float* agg   = state + (size_t)NNODE*H;     // 20000*128 f32

  // bf16 pre-transposed weight cache (~400 KB, L2-resident)
  unsigned short* wbf   = (unsigned short*)(agg + (size_t)NNODE*H);
  unsigned short* msgW1T = wbf;                    // 2 x 16384
  unsigned short* msgW2T = msgW1T + 2*16384;
  unsigned short* attW1T = msgW2T + 2*16384;
  unsigned short* attW2T = attW1T + 2*16384;
  unsigned short* thW1T  = attW2T + 2*16384;
  unsigned short* thW2T  = thW1T + 16384;
  unsigned short* alW1T  = thW2T + 16384;
  unsigned short* alW2T  = alW1T + 16384;
  unsigned short* thW3T  = alW2T + 16384;          // 4096 (padded 32x128)
  unsigned short* alW3T  = thW3T + 4096;

  float* outT = (float*)d_out;
  float* outA = outT + (size_t)EPRED*KMIX;

  const int EDGE_SMEM = 139264;
  const int HEAD_SMEM = 167168;
  (void)hipFuncSetAttribute(reinterpret_cast<const void*>(gran_edge_kernel),
                            hipFuncAttributeMaxDynamicSharedMemorySize, EDGE_SMEM);
  (void)hipFuncSetAttribute(reinterpret_cast<const void*>(gran_head_kernel),
                            hipFuncAttributeMaxDynamicSharedMemorySize, HEAD_SMEM);

  // weight prep (bf16 + transpose), tiny
  for (int l = 0; l < 2; ++l) {
    gran_prepT<<<64, 256, 0, stream>>>(msgW1 + (size_t)l*256*128, msgW1T + l*16384);
    gran_prepT<<<64, 256, 0, stream>>>(msgW2 + (size_t)l*128*128, msgW2T + l*16384);
    gran_prepT<<<64, 256, 0, stream>>>(attW1 + (size_t)l*256*128, attW1T + l*16384);
    gran_prepT<<<64, 256, 0, stream>>>(attW2 + (size_t)l*128*128, attW2T + l*16384);
  }
  gran_prepT<<<64, 256, 0, stream>>>(tW1, thW1T);
  gran_prepT<<<64, 256, 0, stream>>>(tW2, thW2T);
  gran_prepT<<<64, 256, 0, stream>>>(aW1, alW1T);
  gran_prepT<<<64, 256, 0, stream>>>(aW2, alW2T);
  gran_prepT3<<<16, 256, 0, stream>>>(tW3, thW3T);
  gran_prepT3<<<16, 256, 0, stream>>>(aW3, alW3T);

  gran_inproj<<<4096, 128, 0, stream>>>(A_pad, W_in, b_in, nf);
  gran_gather<<<(NNODE*H + 255)/256, 256, 0, stream>>>(nf, nif, state);

  for (int l = 0; l < 2; ++l) {
    if (l) gran_relu<<<(NNODE*H + 255)/256, 256, 0, stream>>>(state, NNODE*H);
    gran_zero<<<(NNODE*H + 255)/256, 256, 0, stream>>>(agg, NNODE*H);
    gran_edge_kernel<<<M_EDGES/(NW*16), 256, EDGE_SMEM, stream>>>(
        state, agg, edges, attIdx,
        msgW1T + l*16384, msgW2T + l*16384, attW1T + l*16384, attW2T + l*16384,
        msgW1 + (size_t)l*256*128, msgb1 + l*128, msgb2 + l*128,
        attW1 + (size_t)l*256*128, attb1 + l*128, attb2 + l*128);
    gran_gru<<<NNODE, 128, 0, stream>>>(agg, state,
        gWih + (size_t)l*128*384, gbih + l*384,
        gWhh + (size_t)l*128*384, gbhh + l*384);
  }

  gran_head_kernel<<<EPRED/(NW*16), 256, HEAD_SMEM, stream>>>(
      state, gnn, thW1T, thW2T, thW3T, alW1T, alW2T, alW3T,
      tb1, tb2, tb3, ab1, ab2, ab3, outT, outA);
  gran_lsm<<<(EPRED + 255)/256, 256, 0, stream>>>(outA, EPRED);
}
